// AdaptiveQuantizer_532575945231
// MI455X (gfx1250) — matrix operand order repaired and checked by reading, not compiled, not run on hardware
//
#include <hip/hip_runtime.h>
#include <hip/hip_bf16.h>

typedef __attribute__((ext_vector_type(16))) _Float16 v16h;
typedef __attribute__((ext_vector_type(8)))  float    v8f;

#define WAVES_PER_BLOCK 8
#define ROWS_PER_WAVE   32
#define ROWS_PER_BLOCK  256
#define HPAD            36   // 32 cols + 4 halves pad to spread LDS banks

__global__ __launch_bounds__(256)
void AdaptiveQuantizer_532575945231_kernel(
    const float* __restrict__ v,      // [B,2]
    const float* __restrict__ snr,    // [B,1]
    const float* __restrict__ gumbel, // [B,3]
    const float* __restrict__ fc1_w,  // [32,3]
    const float* __restrict__ fc1_b,  // [32]
    const float* __restrict__ fc2_w,  // [3,32]
    const float* __restrict__ fc2_b,  // [3]
    const float* __restrict__ s2p,    // [1]
    const float* __restrict__ s4p,    // [1]
    float* __restrict__ out,          // [B*2] vq | [B] eb | [B*3] w
    long nrows)
{
    __shared__ _Float16 h_lds[WAVES_PER_BLOCK][2][16][HPAD];
    __shared__ float    logit_lds[WAVES_PER_BLOCK][32][4];

    const int  tid  = threadIdx.x;
    const int  wave = tid >> 5;
    const int  lane = tid & 31;
    const int  n    = lane & 15;   // N column (B/C/D) or M row (A)
    const int  hi   = lane >> 4;   // lane half
    const long rowBase = (long)blockIdx.x * ROWS_PER_BLOCK + (long)wave * ROWS_PER_WAVE;

    // ---- Build B matrices in WMMA B-layout (constant per kernel) ----
    // B 32x16 f16, k order chosen here (A uses the same): lane -> column N = lane%16; lanes 0-15: element e = K e;
    //              lanes 16-31: element e = K 16+e.
    v16h b1lo = {}, b1hi = {}, b2 = {};
    if (hi == 0) {
        // fc1: B[k][j] = fc1_w[j][k], nonzero only K<3
        int jlo = n, jhi = n + 16;
        b1lo[0] = (_Float16)fc1_w[jlo * 3 + 0];
        b1lo[1] = (_Float16)fc1_w[jlo * 3 + 1];
        b1lo[2] = (_Float16)fc1_w[jlo * 3 + 2];
        b1hi[0] = (_Float16)fc1_w[jhi * 3 + 0];
        b1hi[1] = (_Float16)fc1_w[jhi * 3 + 1];
        b1hi[2] = (_Float16)fc1_w[jhi * 3 + 2];
    }
    if (n < 3) {
        // fc2: B[k][c] = fc2_w[c][k] (row-major [3,32] -> contiguous per column c)
        const int kb = hi ? 16 : 0;
#pragma unroll
        for (int e = 0; e < 16; ++e)
            b2[e] = (_Float16)fc2_w[n * 32 + kb + e];
    }

    // ---- fc1 via WMMA for two 16-row tiles ----
#pragma unroll
    for (int t = 0; t < 2; ++t) {
        // A 16x32 f16: lanes 0-15 hold M=lane, elements 0..7 = K0..7 -> x at K=0,1,2.
        // lanes 16-31 hold K=8..15 / 24..31 -> all zero padding.
        v16h a = {};
        if (hi == 0) {
            long r = rowBase + (long)t * 16 + n;
            if (r >= nrows) r = nrows - 1;           // clamp (B divisible; safety only)
            a[0] = (_Float16)v[2 * r];
            a[1] = (_Float16)v[2 * r + 1];
            a[2] = (_Float16)snr[r];
        }
#pragma unroll
        for (int nh = 0; nh < 2; ++nh) {
            v8f c = {};
            c = __builtin_amdgcn_wmma_f32_16x16x32_f16(
                    false, a, false, (nh == 0) ? b1lo : b1hi,
                    (short)0, c, false, false);
            const int   j    = n + 16 * nh;
            const float bias = fc1_b[j];
#pragma unroll
            for (int k = 0; k < 8; ++k) {
                float hv = c[k] + bias;
                hv = hv > 0.0f ? hv : 0.0f;          // ReLU
                h_lds[wave][t][k + 8 * hi][j] = (_Float16)hv;   // C-layout -> [M][K] in LDS
            }
        }
    }
    __syncthreads();

    // ---- fc2 via WMMA: reload h in A-layout from LDS ----
#pragma unroll
    for (int t = 0; t < 2; ++t) {
        v16h a2;
        const int kb = hi ? 16 : 0;                  // lanes>=16: K=16..31
#pragma unroll
        for (int e = 0; e < 8; ++e) {
            a2[e]     = h_lds[wave][t][n][kb + e];
            a2[8 + e] = h_lds[wave][t][n][kb + 8 + e];
        }
        v8f c = {};
        c = __builtin_amdgcn_wmma_f32_16x16x32_f16(
                false, a2, false, b2, (short)0, c, false, false);
        if (n < 3) {
            const float bias = fc2_b[n];
#pragma unroll
            for (int k = 0; k < 8; ++k)
                logit_lds[wave][t * 16 + k + 8 * hi][n] = c[k] + bias;
        }
    }
    __syncthreads();

    // ---- Elementwise epilogue: one row per lane (full occupancy) ----
    const long r = rowBase + lane;
    if (r < nrows) {
        const float z0 = logit_lds[wave][lane][0] + gumbel[3 * r + 0];
        const float z1 = logit_lds[wave][lane][1] + gumbel[3 * r + 1];
        const float z2 = logit_lds[wave][lane][2] + gumbel[3 * r + 2];
        int   idx  = 0;
        float best = z0;
        if (z1 > best) { best = z1; idx = 1; }       // first-max tie-breaking (jnp.argmax)
        if (z2 > best) { idx = 2; }

        const float s2 = s2p[0];
        const float s4 = s4p[0];
        const float v0 = v[2 * r];
        const float v1 = v[2 * r + 1];

        float vq0 = 0.0f, vq1 = 0.0f, eb = 0.0f;
        if (idx == 1) {            // 2-bit: qn=-2, qp=1
            vq0 = rintf(fminf(fmaxf(v0 / s2, -2.0f), 1.0f)) * s2;
            vq1 = rintf(fminf(fmaxf(v1 / s2, -2.0f), 1.0f)) * s2;
            eb  = 2.0f;
        } else if (idx == 2) {     // 4-bit: qn=-8, qp=7
            vq0 = rintf(fminf(fmaxf(v0 / s4, -8.0f), 7.0f)) * s4;
            vq1 = rintf(fminf(fmaxf(v1 / s4, -8.0f), 7.0f)) * s4;
            eb  = 4.0f;
        }

        out[2 * r]     = vq0;
        out[2 * r + 1] = vq1;
        out[2 * nrows + r] = eb;
        float* w = out + 3 * nrows;
        w[3 * r + 0] = (idx == 0) ? 1.0f : 0.0f;
        w[3 * r + 1] = (idx == 1) ? 1.0f : 0.0f;
        w[3 * r + 2] = (idx == 2) ? 1.0f : 0.0f;
    }
}

extern "C" void kernel_launch(void* const* d_in, const int* in_sizes, int n_in,
                              void* d_out, int out_size, void* d_ws, size_t ws_size,
                              hipStream_t stream) {
    const float* v      = (const float*)d_in[0];
    const float* snr    = (const float*)d_in[1];
    const float* gumbel = (const float*)d_in[2];
    const float* fc1_w  = (const float*)d_in[3];
    const float* fc1_b  = (const float*)d_in[4];
    const float* fc2_w  = (const float*)d_in[5];
    const float* fc2_b  = (const float*)d_in[6];
    const float* s2     = (const float*)d_in[7];
    const float* s4     = (const float*)d_in[8];
    float*       out    = (float*)d_out;

    const long nrows  = (long)in_sizes[0] / 2;   // v is [B,2]
    const int  blocks = (int)((nrows + ROWS_PER_BLOCK - 1) / ROWS_PER_BLOCK);

    AdaptiveQuantizer_532575945231_kernel<<<blocks, 256, 0, stream>>>(
        v, snr, gumbel, fc1_w, fc1_b, fc2_w, fc2_b, s2, s4, out, nrows);
}
